// OctreeMamba_4286377361902
// MI455X (gfx1250) — matrix operand order repaired and checked by reading, not compiled, not run on hardware
//
#include <hip/hip_runtime.h>
#include <hip/hip_bf16.h>

// ---------------- constants (from reference) ----------------
#define N_TOK 32768
#define DM    512
#define DI    1024
#define DS    16
#define DR    32
#define DC    4

typedef __bf16        v16bf __attribute__((ext_vector_type(16)));
typedef float         v8f   __attribute__((ext_vector_type(8)));
typedef unsigned int  v4u   __attribute__((ext_vector_type(4)));

using bf16 = __hip_bfloat16;

union FragAB { v16bf v; v4u q[2]; };

__device__ __forceinline__ float bf2f(bf16 x)  { return __bfloat162float(x); }
__device__ __forceinline__ bf16  f2bf(float x) { return __float2bfloat16(x); }
__device__ __forceinline__ float dsilu(float x){ return x / (1.0f + __expf(-x)); }

// ---------------------------------------------------------------------------
// bf16 WMMA GEMM with LDS double buffering:
//   out[M x Kout] = A[M x Kin] * W[Kout x Kin]^T
// Block tile: 128 (M) x 64 (N); 8 waves, each wave: 4 M-tiles x 1 N-tile.
// K is consumed in CHUNK-sized slices staged through LDS; global loads for
// chunk c+1 are issued before the WMMAs of chunk c.  All 10 fragment
// ds_loads of a k-step are batched ahead of its 4 back-to-back WMMAs
// (disjoint accumulators -> no data hazard, one dscnt wait per k-step).
// OUT_BF16: 1 -> bf16 output, 0 -> f32 output.  ACT: 0 none, 1 softplus.
// ---------------------------------------------------------------------------
template<int OUT_BF16, int ACT, int CHUNK>
__global__ void __launch_bounds__(256)
gemm_lds_wmma(const bf16* __restrict__ A, int lda,
              const bf16* __restrict__ W, int Kin,
              void* __restrict__ outp, int ldo,
              const float* __restrict__ bias)
{
    constexpr int APITCH = CHUNK + 8;   // +8 elems pad -> conflict-free rows
    constexpr int BPITCH = CHUNK + 8;
    __shared__ __align__(16) bf16 lA[2][128 * APITCH];
    __shared__ __align__(16) bf16 lB[2][64 * BPITCH];

    const int tid   = threadIdx.x;
    const int lane  = tid & 31;
    const int wv    = tid >> 5;
    const int hl    = lane >> 4;          // half-wave: 0 or 1
    const int l16   = lane & 15;
    const int mblk  = blockIdx.x * 128;
    const int nblk  = blockIdx.y * 64;
    const int mwave = (wv >> 2) * 64;     // wave's M base inside block
    const int ntile = (wv & 3);           // wave's N tile inside block

    // global->reg staging, 16-byte units spread over 256 threads
    constexpr int UPR = CHUNK / 8;                       // 16B units per row
    constexpr int AUN = (128 * UPR) / 256;               // A units per thread
    constexpr int BUN = (64  * UPR) / 256;               // B units per thread
    v4u ra[AUN], rb[BUN];
    int arow[AUN], acol[AUN], brow[BUN], bcol[BUN];
#pragma unroll
    for (int q = 0; q < AUN; ++q) {
        const int u = tid * AUN + q;
        arow[q] = u / UPR;
        acol[q] = (u % UPR) * 8;
    }
#pragma unroll
    for (int q = 0; q < BUN; ++q) {
        const int u = tid * BUN + q;
        brow[q] = u / UPR;
        bcol[q] = (u % UPR) * 8;
    }

    auto gload = [&](int k0) {
#pragma unroll
        for (int q = 0; q < AUN; ++q)
            ra[q] = *(const v4u*)(A + (size_t)(mblk + arow[q]) * lda + k0 + acol[q]);
#pragma unroll
        for (int q = 0; q < BUN; ++q)
            rb[q] = *(const v4u*)(W + (size_t)(nblk + brow[q]) * Kin + k0 + bcol[q]);
    };
    auto lstore = [&](int buf) {
#pragma unroll
        for (int q = 0; q < AUN; ++q)
            *(v4u*)&lA[buf][arow[q] * APITCH + acol[q]] = ra[q];
#pragma unroll
        for (int q = 0; q < BUN; ++q)
            *(v4u*)&lB[buf][brow[q] * BPITCH + bcol[q]] = rb[q];
    };

    v8f acc[4] = {};

    gload(0);
    lstore(0);
    __syncthreads();

    const int nchunks = Kin / CHUNK;
    for (int c = 0; c < nchunks; ++c) {
        const int buf = c & 1;
        if (c + 1 < nchunks) gload((c + 1) * CHUNK);   // overlap with WMMAs

#pragma unroll
        for (int kk = 0; kk < CHUNK; kk += 32) {
            // Batch ALL fragment loads of this k-step, then 4 WMMAs.
            // B fragment (32x16): lane l16 = column; lanes 0-15 K=kk..kk+15,
            // lanes 16-31 K=kk+16..kk+31
            FragAB bfr, afr[4];
            {
                const bf16* bp = &lB[buf][(ntile * 16 + l16) * BPITCH + kk + hl * 16];
                bfr.q[0] = *(const v4u*)(bp);
                bfr.q[1] = *(const v4u*)(bp + 8);
            }
#pragma unroll
            for (int i = 0; i < 4; ++i) {
                // A fragment (16x32): lane = row; lanes 0-15 K=kk..kk+15,
                // lanes 16-31 K=kk+16..kk+31
                const bf16* ap = &lA[buf][(mwave + i * 16 + l16) * APITCH + kk + hl * 16];
                afr[i].q[0] = *(const v4u*)(ap);
                afr[i].q[1] = *(const v4u*)(ap + 8);
            }
#pragma unroll
            for (int i = 0; i < 4; ++i) {
                acc[i] = __builtin_amdgcn_wmma_f32_16x16x32_bf16(
                             false, afr[i].v, false, bfr.v, (short)0, acc[i], false, false);
            }
        }
        __syncthreads();                       // all reads of lds[buf] done
        if (c + 1 < nchunks) {
            lstore(buf ^ 1);                   // loadcnt wait lands here
            __syncthreads();                   // writes visible before next reads
        }
    }

    // C/D layout: VGPR r -> M = r + 8*(lane>=16), N = lane&15
#pragma unroll
    for (int i = 0; i < 4; ++i) {
#pragma unroll
        for (int r = 0; r < 8; ++r) {
            const int row = mblk + mwave + i * 16 + hl * 8 + r;
            const int col = nblk + ntile * 16 + l16;
            float v = acc[i][r];
            if (bias) v += bias[col];
            if (ACT == 1) v = (v > 20.0f) ? v : log1pf(__expf(v));   // softplus
            const size_t o = (size_t)row * ldo + col;
            if (OUT_BF16) ((bf16*)outp)[o] = f2bf(v);
            else          ((float*)outp)[o] = v;
        }
    }
}

// ---------------------------------------------------------------------------
// RMSNorm: one block (128 threads) per row of 512, bf16 output h = x*rms*w
// ---------------------------------------------------------------------------
__global__ void __launch_bounds__(128)
rmsnorm_k(const float* __restrict__ x, const float* __restrict__ w,
          bf16* __restrict__ out)
{
    __shared__ float red[4];
    const int row = blockIdx.x, tid = threadIdx.x;
    const float* xr = x + (size_t)row * DM;
    float v[4], ss = 0.f;
#pragma unroll
    for (int k = 0; k < 4; ++k) { v[k] = xr[tid + 128 * k]; ss += v[k] * v[k]; }
#pragma unroll
    for (int m = 16; m >= 1; m >>= 1) ss += __shfl_xor(ss, m, 32);
    if ((tid & 31) == 0) red[tid >> 5] = ss;
    __syncthreads();
    const float sc = rsqrtf((red[0] + red[1] + red[2] + red[3]) / (float)DM + 1e-5f);
#pragma unroll
    for (int k = 0; k < 4; ++k) {
        const int c = tid + 128 * k;
        out[(size_t)row * DM + c] = f2bf(v[k] * sc * w[c]);
    }
}

// out = rmsnorm(h2 + x0, w) - x0   (bf16 out, feeds the final projection GEMM)
__global__ void __launch_bounds__(128)
resnorm_k(const float* __restrict__ h2, const float* __restrict__ x0,
          const float* __restrict__ w, bf16* __restrict__ out)
{
    __shared__ float red[4];
    const int row = blockIdx.x, tid = threadIdx.x;
    float xv[4], v[4], ss = 0.f;
#pragma unroll
    for (int k = 0; k < 4; ++k) {
        const int c = tid + 128 * k;
        xv[k] = x0[(size_t)row * DM + c];
        v[k]  = h2[(size_t)row * DM + c] + xv[k];
        ss += v[k] * v[k];
    }
#pragma unroll
    for (int m = 16; m >= 1; m >>= 1) ss += __shfl_xor(ss, m, 32);
    if ((tid & 31) == 0) red[tid >> 5] = ss;
    __syncthreads();
    const float sc = rsqrtf((red[0] + red[1] + red[2] + red[3]) / (float)DM + 1e-5f);
#pragma unroll
    for (int k = 0; k < 4; ++k) {
        const int c = tid + 128 * k;
        out[(size_t)row * DM + c] = f2bf(v[k] * sc * w[c] - xv[k]);
    }
}

// ---------------------------------------------------------------------------
// Depthwise causal conv (DC=4) + SiLU.  BWD=1 operates on the reversed
// sequence (writes into the reversed domain).  x lives in xz[:, :DI] (bf16).
// ---------------------------------------------------------------------------
template<int BWD>
__global__ void __launch_bounds__(256)
conv_silu_k(const bf16* __restrict__ xz, const float* __restrict__ cw,
            const float* __restrict__ cb, bf16* __restrict__ out)
{
    const size_t idx = (size_t)blockIdx.x * 256 + threadIdx.x;  // over N*DI
    const int d = (int)(idx & (DI - 1));
    const int t = (int)(idx >> 10);
    float acc = cb[d];
#pragma unroll
    for (int j = 0; j < DC; ++j) {
        const int s = t - (DC - 1) + j;
        if (s >= 0) {
            const int src = BWD ? (N_TOK - 1 - s) : s;
            acc += bf2f(xz[(size_t)src * (2 * DI) + d]) * cw[d * DC + j];
        }
    }
    out[idx] = f2bf(dsilu(acc));
}

// ---------------------------------------------------------------------------
// Selective scan.  One wave = 2 channels x 16 states (lane = 16*ch + state).
// h_t = exp(dt*A)*h + dt*u*B_t ;  y_t = sum_n h*C_n, reduced over 16 lanes.
// Gated by silu(z) and combined with D*u.  BWD accumulates into yacc.
// ---------------------------------------------------------------------------
template<int BWD>
__global__ void __launch_bounds__(256)
scan_k(const float* __restrict__ dt, const bf16* __restrict__ xc,
       const bf16* __restrict__ xdbl, const float* __restrict__ A_log,
       const float* __restrict__ Dp, const bf16* __restrict__ xz,
       float* __restrict__ yacc)
{
    const int lane = threadIdx.x & 31;
    const int n    = lane & 15;
    const int d    = blockIdx.x * 16 + (threadIdx.x >> 5) * 2 + (lane >> 4);
    const float a  = -__expf(A_log[d * DS + n]);
    const float Dd = Dp[d];
    float h = 0.f;

    for (int t = 0; t < N_TOK; ++t) {
        if (((t & 15) == 0) && (t + 32 < N_TOK)) {
            __builtin_prefetch((const void*)(dt + (size_t)(t + 32) * DI + d), 0, 0);
            __builtin_prefetch((const void*)(xc + (size_t)(t + 32) * DI + d), 0, 0);
        }
        const float dtv = dt[(size_t)t * DI + d];
        const float u   = bf2f(xc[(size_t)t * DI + d]);
        const float Bv  = bf2f(xdbl[(size_t)t * 64 + DR + n]);
        const float Cv  = bf2f(xdbl[(size_t)t * 64 + DR + DS + n]);
        h = __expf(dtv * a) * h + (dtv * u) * Bv;
        float s = h * Cv;
        s += __shfl_xor(s, 1, 16);
        s += __shfl_xor(s, 2, 16);
        s += __shfl_xor(s, 4, 16);
        s += __shfl_xor(s, 8, 16);
        if (n == 0) {
            const int tt = BWD ? (N_TOK - 1 - t) : t;
            const float zv  = bf2f(xz[(size_t)tt * (2 * DI) + DI + d]);
            const float val = (s + Dd * u) * dsilu(zv);
            if (BWD) yacc[(size_t)tt * DI + d] += val;
            else     yacc[(size_t)tt * DI + d]  = val;
        }
    }
}

// ---------------------------------------------------------------------------
__global__ void __launch_bounds__(256)
f32_to_bf16_k(const float* __restrict__ in, bf16* __restrict__ out, long long n)
{
    const long long i = (long long)blockIdx.x * 256 + threadIdx.x;
    if (i < n) out[i] = f2bf(in[i]);
}

// ---------------------------------------------------------------------------
extern "C" void kernel_launch(void* const* d_in, const int* in_sizes, int n_in,
                              void* d_out, int out_size, void* d_ws, size_t ws_size,
                              hipStream_t stream)
{
    (void)in_sizes; (void)n_in; (void)out_size; (void)ws_size;

    const float* data       = (const float*)d_in[0];
    const float* w_norm     = (const float*)d_in[2];
    const float* in_proj_w  = (const float*)d_in[3];
    const float* conv_w     = (const float*)d_in[4];
    const float* conv_b     = (const float*)d_in[5];
    const float* x_proj_w   = (const float*)d_in[6];
    const float* dt_proj_w  = (const float*)d_in[7];
    const float* dt_proj_b  = (const float*)d_in[8];
    const float* A_log      = (const float*)d_in[9];
    const float* Dp         = (const float*)d_in[10];
    const float* conv_w_b   = (const float*)d_in[11];
    const float* conv_b_b   = (const float*)d_in[12];
    const float* x_proj_w_b = (const float*)d_in[13];
    const float* dt_proj_w_b= (const float*)d_in[14];
    const float* dt_proj_b_b= (const float*)d_in[15];
    const float* A_b_log    = (const float*)d_in[16];
    const float* D_b        = (const float*)d_in[17];
    const float* out_proj_w = (const float*)d_in[18];
    const float* w_norm_f   = (const float*)d_in[19];
    const float* proj_w     = (const float*)d_in[20];
    const float* proj_b     = (const float*)d_in[21];

    // ---- workspace carve-up ----
    char* ws = (char*)d_ws;
    size_t off = 0;
    auto carve = [&](size_t bytes) -> void* {
        void* p = ws + off;
        off = (off + bytes + 255) & ~(size_t)255;
        return p;
    };
    bf16*  hbf   = (bf16*) carve((size_t)N_TOK * DM * 2);        // normed input, bf16
    bf16*  xz    = (bf16*) carve((size_t)N_TOK * 2 * DI * 2);    // in_proj output
    bf16*  xcf   = (bf16*) carve((size_t)N_TOK * DI * 2);        // conv+silu fwd
    bf16*  xcb   = (bf16*) carve((size_t)N_TOK * DI * 2);        // conv+silu bwd (reversed domain)
    bf16*  xdf   = (bf16*) carve((size_t)N_TOK * 64 * 2);        // x_dbl fwd
    bf16*  xdb   = (bf16*) carve((size_t)N_TOK * 64 * 2);        // x_dbl bwd
    float* dtf   = (float*)carve((size_t)N_TOK * DI * 4);        // softplus dt fwd
    float* dtb   = (float*)carve((size_t)N_TOK * DI * 4);        // softplus dt bwd
    float* yacc  = (float*)carve((size_t)N_TOK * DI * 4);        // y_f + y_b (gated)
    bf16*  ybf   = (bf16*) carve((size_t)N_TOK * DI * 2);
    float* h2    = (float*)carve((size_t)N_TOK * DM * 4);        // out_proj result
    bf16*  nbf   = (bf16*) carve((size_t)N_TOK * DM * 2);        // final normed, bf16
    bf16*  w_in  = (bf16*) carve((size_t)2 * DI * DM * 2);
    bf16*  w_xp  = (bf16*) carve((size_t)64 * DI * 2);
    bf16*  w_xpb = (bf16*) carve((size_t)64 * DI * 2);
    bf16*  w_dt  = (bf16*) carve((size_t)DI * DR * 2);
    bf16*  w_dtb = (bf16*) carve((size_t)DI * DR * 2);
    bf16*  w_out = (bf16*) carve((size_t)DM * DI * 2);
    bf16*  w_prj = (bf16*) carve((size_t)DM * DM * 2);

    auto cvt = [&](const float* src, bf16* dst, long long n) {
        f32_to_bf16_k<<<(unsigned)((n + 255) / 256), 256, 0, stream>>>(src, dst, n);
    };

    // ---- weight conversion to bf16 ----
    cvt(in_proj_w,   w_in,  (long long)2 * DI * DM);
    cvt(x_proj_w,    w_xp,  (long long)64 * DI);
    cvt(x_proj_w_b,  w_xpb, (long long)64 * DI);
    cvt(dt_proj_w,   w_dt,  (long long)DI * DR);
    cvt(dt_proj_w_b, w_dtb, (long long)DI * DR);
    cvt(out_proj_w,  w_out, (long long)DM * DI);
    cvt(proj_w,      w_prj, (long long)DM * DM);

    // ---- 1. rmsnorm ----
    rmsnorm_k<<<N_TOK, 128, 0, stream>>>(data, w_norm, hbf);

    // ---- 2. in_proj: [N,512] x [2048,512]^T -> xz bf16 ----
    gemm_lds_wmma<1, 0, 64><<<dim3(N_TOK / 128, (2 * DI) / 64), 256, 0, stream>>>(
        hbf, DM, w_in, DM, xz, 2 * DI, nullptr);

    // ---- 3. causal conv + silu (both directions) ----
    conv_silu_k<0><<<(N_TOK * DI) / 256, 256, 0, stream>>>(xz, conv_w,   conv_b,   xcf);
    conv_silu_k<1><<<(N_TOK * DI) / 256, 256, 0, stream>>>(xz, conv_w_b, conv_b_b, xcb);

    // ---- 4. x_proj: [N,1024] x [64,1024]^T -> x_dbl ----
    gemm_lds_wmma<1, 0, 64><<<dim3(N_TOK / 128, 1), 256, 0, stream>>>(
        xcf, DI, w_xp, DI, xdf, 64, nullptr);
    gemm_lds_wmma<1, 0, 64><<<dim3(N_TOK / 128, 1), 256, 0, stream>>>(
        xcb, DI, w_xpb, DI, xdb, 64, nullptr);

    // ---- 5. dt_proj + softplus: [N,32] x [1024,32]^T -> dt f32 ----
    gemm_lds_wmma<0, 1, 32><<<dim3(N_TOK / 128, DI / 64), 256, 0, stream>>>(
        xdf, 64, w_dt, DR, dtf, DI, dt_proj_b);
    gemm_lds_wmma<0, 1, 32><<<dim3(N_TOK / 128, DI / 64), 256, 0, stream>>>(
        xdb, 64, w_dtb, DR, dtb, DI, dt_proj_b_b);

    // ---- 6. selective scan (fwd writes, bwd accumulates) ----
    scan_k<0><<<DI / 16, 256, 0, stream>>>(dtf, xcf, xdf, A_log,   Dp,  xz, yacc);
    scan_k<1><<<DI / 16, 256, 0, stream>>>(dtb, xcb, xdb, A_b_log, D_b, xz, yacc);

    // ---- 7. out_proj: [N,1024] x [512,1024]^T -> h2 f32 ----
    cvt(yacc, ybf, (long long)N_TOK * DI);
    gemm_lds_wmma<0, 0, 64><<<dim3(N_TOK / 128, DM / 64), 256, 0, stream>>>(
        ybf, DI, w_out, DI, h2, DM, nullptr);

    // ---- 8. residual + rmsnorm - x0 ----
    resnorm_k<<<N_TOK, 128, 0, stream>>>(h2, data, w_norm_f, nbf);

    // ---- 9. final proj + bias -> d_out f32 ----
    gemm_lds_wmma<0, 0, 64><<<dim3(N_TOK / 128, DM / 64), 256, 0, stream>>>(
        nbf, DM, w_prj, DM, (float*)d_out, DM, proj_b);
}
